// RelationalFeatureExtractor_38646115729916
// MI455X (gfx1250) — matrix operand order repaired and checked by reading, not compiled, not run on hardware
//
#include <hip/hip_runtime.h>

typedef __attribute__((ext_vector_type(16))) __bf16       v16bf;
typedef __attribute__((ext_vector_type(2)))  __bf16       v2bf;
typedef __attribute__((ext_vector_type(8)))  float        v8f;
typedef __attribute__((ext_vector_type(8)))  unsigned int v8u;
typedef __attribute__((ext_vector_type(4)))  float        f4;

#define NN_    65536          // N*N
#define TOTAL  (16 * 65536)   // B*N*N
#define HID_   64
#define HEADS_ 8
#define LOG2E_ 1.4426950408889634f

static __device__ __forceinline__ v8u zero8u() {
  v8u u = {0u, 0u, 0u, 0u, 0u, 0u, 0u, 0u};
  return u;
}

// f32 -> packed bf16 pair (low = even k, high = odd k)
static __device__ __forceinline__ unsigned bfpack(float a, float b) {
#if __has_builtin(__builtin_amdgcn_cvt_pk_bf16_f32)
  v2bf p = __builtin_amdgcn_cvt_pk_bf16_f32(a, b);
  return __builtin_bit_cast(unsigned, p);
#else
  unsigned ua = __builtin_bit_cast(unsigned, a) + 0x8000u;
  unsigned ub = __builtin_bit_cast(unsigned, b) + 0x8000u;
  // result = { ub[31:16], ua[31:16] }
  return __builtin_amdgcn_perm(ub, ua, 0x07060302u);
#endif
}

static __device__ __forceinline__ float rcpf(float x) {
  return __builtin_amdgcn_rcpf(x);
}
static __device__ __forceinline__ float exp2f_hw(float x) {
  return __builtin_amdgcn_exp2f(x);
}

static __device__ __forceinline__ v8f wmma_bf(v16bf a, v16bf b, v8f c) {
  return __builtin_amdgcn_wmma_f32_16x16x32_bf16(false, a, false, b, (short)0, c,
                                                 false, false);
}

__global__ __launch_bounds__(256) void relnet_kernel(
    const float* __restrict__ ff, const float* __restrict__ W1,
    const float* __restrict__ b1, const float* __restrict__ W2,
    const float* __restrict__ b2, float* __restrict__ out) {
  const int  lane = threadIdx.x & 31;
  const int  col  = lane & 15;
  const bool lo   = lane < 16;
  const int  half = lane >> 4;
  const int  wave = blockIdx.x * (blockDim.x >> 5) + (threadIdx.x >> 5);
  const int  nw   = gridDim.x * (blockDim.x >> 5);

  // ---- A fragments: W1^T, 4 tiles of (16 hid) x (K=12 padded to 32) ----
  v16bf aw1[4];
#pragma unroll
  for (int t = 0; t < 4; ++t) {
    v8u       u   = zero8u();
    const int hid = col + 16 * t;
    if (lo) {  // lanes 0-15: K = 0..11 (12..15 zero pad); lanes 16-31: K 16..31 = 0
#pragma unroll
      for (int i = 0; i < 6; ++i)
        u[i] = bfpack(W1[(2 * i) * HID_ + hid], W1[(2 * i + 1) * HID_ + hid]);
    }
    aw1[t] = __builtin_bit_cast(v16bf, u);
  }

  // ---- A fragments: W2^T, (16 x 64) with rows >= 8 zero, two K=32 chunks ----
  v16bf aw2[2];
#pragma unroll
  for (int j = 0; j < 2; ++j) {
    v8u u = zero8u();
    if (col < 8) {
      const int kb = 32 * j + (lo ? 0 : 16);
#pragma unroll
      for (int i = 0; i < 8; ++i)
        u[i] = bfpack(W2[(kb + 2 * i) * HEADS_ + col],
                      W2[(kb + 2 * i + 1) * HEADS_ + col]);
    }
    aw2[j] = __builtin_bit_cast(v16bf, u);
  }

  // ---- C inits: b1 varies along M (= hid), b2 along M (= head) ----
  v8f cb1[4];
#pragma unroll
  for (int t = 0; t < 4; ++t)
#pragma unroll
    for (int r = 0; r < 8; ++r) cb1[t][r] = b1[16 * t + 8 * half + r];
  v8f cb2;
#pragma unroll
  for (int r = 0; r < 8; ++r) cb2[r] = lo ? b2[r] : 0.0f;

  for (int base = wave * 32; base < TOTAL; base += nw * 32) {
    const int q = base + lane;
    f4        v = *(const f4*)(ff + (size_t)q * 4);
    float rx = v[0], ry = v[1], rvx = v[2], rvy = v[3];

    float dist      = __builtin_amdgcn_sqrtf(rx * rx + ry * ry + 1e-6f);
    float inv_d     = rcpf(dist + 1e-6f);
    float inv_dist  = rcpf(dist + 0.1f);
    float speed_sq  = rvx * rvx + rvy * rvy;
    float rel_speed = __builtin_amdgcn_sqrtf(speed_sq + 1e-6f);
    float dot_vp    = rvx * rx + rvy * ry;
    float closing   = dot_vp * inv_d;
    float tx        = -dot_vp * rcpf(speed_sq + 1e-6f);
    tx              = fminf(fmaxf(tx, -15.0f), 15.0f);  // tanh saturation guard
    float e2        = exp2f_hw((2.0f * LOG2E_) * tx);   // e^(2*tx)
    float ttca      = (e2 - 1.0f) * rcpf(e2 + 1.0f);
    float dirx = rx * inv_d, diry = ry * inv_d;

    // per-lane packed feature column (12 values + pad)
    v8u fp = zero8u();
    fp[0]  = bfpack(rx, ry);
    fp[1]  = bfpack(rvx, rvy);
    fp[2]  = bfpack(dist, inv_dist);
    fp[3]  = bfpack(rel_speed, closing);
    fp[4]  = bfpack(dirx, diry);
    fp[5]  = bfpack(ttca, dot_vp);

    v8u fsw = zero8u();
#pragma unroll
    for (int i = 0; i < 6; ++i) fsw[i] = (unsigned)__shfl_xor((int)fp[i], 16);

#pragma unroll
    for (int p = 0; p < 2; ++p) {
      // B fragment of F^T: cols = 16 pairs, K = 12 (lanes>=16 hold K 16..31 = 0)
      v8u bu = zero8u();
      if (lo) {
#pragma unroll
        for (int i = 0; i < 6; ++i) bu[i] = p ? fsw[i] : fp[i];
      }
      v16bf bfeat = __builtin_bit_cast(v16bf, bu);

      // GEMM1: h^T tiles = W1^T x F^T + b1
      v8f c0 = wmma_bf(aw1[0], bfeat, cb1[0]);
      v8f c1 = wmma_bf(aw1[1], bfeat, cb1[1]);
      v8f c2 = wmma_bf(aw1[2], bfeat, cb1[2]);
      v8f c3 = wmma_bf(aw1[3], bfeat, cb1[3]);

      // SiLU: x * rcp(1 + exp2(-x*log2e))  -- 2 trans + 3 valu per element
#pragma unroll
      for (int r = 0; r < 8; ++r) {
        c0[r] = c0[r] * rcpf(1.0f + exp2f_hw(c0[r] * -LOG2E_));
        c1[r] = c1[r] * rcpf(1.0f + exp2f_hw(c1[r] * -LOG2E_));
        c2[r] = c2[r] * rcpf(1.0f + exp2f_hw(c2[r] * -LOG2E_));
        c3[r] = c3[r] * rcpf(1.0f + exp2f_hw(c3[r] * -LOG2E_));
      }

      // pack h to bf16 pairs; half-wave swaps build B fragments of h (64x16)
      unsigned pk0[4], pk1[4], pk2[4], pk3[4];
#pragma unroll
      for (int i = 0; i < 4; ++i) {
        pk0[i] = bfpack(c0[2 * i], c0[2 * i + 1]);
        pk1[i] = bfpack(c1[2 * i], c1[2 * i + 1]);
        pk2[i] = bfpack(c2[2 * i], c2[2 * i + 1]);
        pk3[i] = bfpack(c3[2 * i], c3[2 * i + 1]);
      }
      v8u u0, u1;
#pragma unroll
      for (int i = 0; i < 4; ++i) {
        unsigned s0 = (unsigned)__shfl_xor((int)pk0[i], 16);
        unsigned s1 = (unsigned)__shfl_xor((int)pk1[i], 16);
        unsigned s2 = (unsigned)__shfl_xor((int)pk2[i], 16);
        unsigned s3 = (unsigned)__shfl_xor((int)pk3[i], 16);
        u0[i]     = lo ? pk0[i] : s1;   // K 0..7   | K 16..23
        u0[i + 4] = lo ? s0 : pk1[i];   // K 8..15  | K 24..31
        u1[i]     = lo ? pk2[i] : s3;   // K 32..39 | K 48..55
        u1[i + 4] = lo ? s2 : pk3[i];   // K 40..47 | K 56..63
      }

      // GEMM2: D = W2^T x h + b2   (M = head, cols = pairs)
      v8f d = wmma_bf(aw2[0], __builtin_bit_cast(v16bf, u0), cb2);
      d     = wmma_bf(aw2[1], __builtin_bit_cast(v16bf, u1), d);

      // store: out[(b*8 + o)*65536 + (q mod 65536)], coalesced along pairs
      const int q0 = base + 16 * p;
      const int qq = (q0 & (NN_ - 1)) + col;
      const int bb = q0 >> 16;
#pragma unroll
      for (int i = 0; i < 4; ++i) {
        float hv = __shfl_xor(d[i + 4], 16);   // hi lanes take o = i+4
        float sv = lo ? d[i] : hv;
        int   o  = lo ? i : (i + 4);
        out[(size_t)(bb * 8 + o) * NN_ + qq] = sv;
      }
    }
  }
}

extern "C" void kernel_launch(void* const* d_in, const int* in_sizes, int n_in,
                              void* d_out, int out_size, void* d_ws,
                              size_t ws_size, hipStream_t stream) {
  const float* ff = (const float*)d_in[0];
  const float* W1 = (const float*)d_in[1];
  const float* b1 = (const float*)d_in[2];
  const float* W2 = (const float*)d_in[3];
  const float* b2 = (const float*)d_in[4];
  float*       out = (float*)d_out;
  // 512 blocks x 256 threads = 4096 waves; 32 pairs/wave/iter; 8 iters exactly.
  relnet_kernel<<<dim3(512), dim3(256), 0, stream>>>(ff, W1, b1, W2, b2, out);
}
